// SimpleLSTM_19490561590060
// MI455X (gfx1250) — compile-verified, matrix operand order checked by reading, not run on hardware
//
#include <hip/hip_runtime.h>

// Persistent fused LSTM for MI455X (gfx1250, wave32).
// B=256 batch split into 16 workgroups of 16 rows; 8 waves per workgroup each
// own 16 of the 128 gate columns. K = H = 32 => whole recurrent K-reduction is
// a single v_wmma_f32_16x16x32_f16 per tile per timestep (minimal dependent
// chain on the latency-bound recurrence). Weights live in registers for the
// whole T loop; cell state c lives in registers; h is exchanged via LDS.
// Activations use raw v_rcp_f32 (no IEEE divide sequences on the chain).

typedef __attribute__((ext_vector_type(16))) _Float16 v16h;
typedef __attribute__((ext_vector_type(8)))  _Float16 v8h;
typedef __attribute__((ext_vector_type(8)))  float    v8f;
typedef __attribute__((ext_vector_type(4)))  float    v4f;

constexpr int kH       = 32;    // hidden size
constexpr int kG       = 128;   // 4*H gate width
constexpr int kMTile   = 16;    // batch rows per workgroup
constexpr int kThreads = 256;   // 8 waves

__device__ __forceinline__ float fast_rcp(float x) {
    return __builtin_amdgcn_rcpf(x);        // raw v_rcp_f32, no NR refinement
}
__device__ __forceinline__ float sigmoid_f(float x) {
    // 1/(1+e^-x); exp overflow -> inf -> rcp(inf) = 0 (safe saturation)
    return fast_rcp(1.0f + __expf(-x));
}
__device__ __forceinline__ float tanh_f(float x) {
    // 1 - 2/(1+e^{2x}); saturates cleanly at +/-1 without NaNs
    return 1.0f - 2.0f * fast_rcp(1.0f + __expf(2.0f * x));
}

__device__ __forceinline__ v16h cvt16(v4f a, v4f b, v4f c, v4f d) {
    v16h r;
    r[0]  = (_Float16)a[0]; r[1]  = (_Float16)a[1]; r[2]  = (_Float16)a[2]; r[3]  = (_Float16)a[3];
    r[4]  = (_Float16)b[0]; r[5]  = (_Float16)b[1]; r[6]  = (_Float16)b[2]; r[7]  = (_Float16)b[3];
    r[8]  = (_Float16)c[0]; r[9]  = (_Float16)c[1]; r[10] = (_Float16)c[2]; r[11] = (_Float16)c[3];
    r[12] = (_Float16)d[0]; r[13] = (_Float16)d[1]; r[14] = (_Float16)d[2]; r[15] = (_Float16)d[3];
    return r;
}

// A/B 16-bit WMMA lane layout over K=32: lane<16 holds K={0..7,16..23} of its
// row/col, lane>=16 holds K={8..15,24..31}. `p` points at row_start + kbase,
// so we read chunks [0..7] and [16..23] relative to p.
__device__ __forceinline__ v16h load_tile_row_f32(const float* __restrict__ p) {
    v4f a = *(const v4f*)(p);
    v4f b = *(const v4f*)(p + 4);
    v4f c = *(const v4f*)(p + 16);
    v4f d = *(const v4f*)(p + 20);
    return cvt16(a, b, c, d);
}

__global__ __launch_bounds__(kThreads)
void lstm_persistent_wmma(const float* __restrict__ x,
                          const float* __restrict__ Wih,
                          const float* __restrict__ Whh,
                          const float* __restrict__ bih,
                          const float* __restrict__ bhh,
                          float* __restrict__ out,
                          int T) {
    __shared__ float Gls[kMTile * kG];                    // raw gates, 8 KB
    __shared__ alignas(16) _Float16 hls[kMTile * kH];     // h state (f16), 1 KB

    const int tid  = threadIdx.x;
    const int lane = tid & 31;
    const int wave = tid >> 5;
    const int n0   = wave * 16;              // this wave's gate-column tile
    const int b0   = blockIdx.x * kMTile;    // this block's batch-row tile
    const int col  = lane & 15;              // tile-local row (A) / col (B,C)
    const int kb   = (lane < 16) ? 0 : 8;    // K base for A/B layout
    const int grow = (lane >> 4) * 8;        // C-layout row base

    // --- one-time: pin this wave's B-tiles of W_ih^T / W_hh^T in registers ---
    const int wrow = n0 + col;                         // gate row of W (= col of W^T)
    const v16h Bih = load_tile_row_f32(Wih + wrow * kH + kb);
    const v16h Bhh = load_tile_row_f32(Whh + wrow * kH + kb);
    const float bias = bih[wrow] + bhh[wrow];          // fused biases, per C-col
    // bias splat as WMMA C init: removes 8 dependent v_add_f32 per step
    v8f cbias;
#pragma unroll
    for (int v = 0; v < 8; ++v) cbias[v] = bias;

    // --- x prefetch pipeline: lane covers batch row b0+col, A layout ---
    const float* xp = x + ((size_t)(b0 + col) * (size_t)T) * kH + kb;
    v4f xA = *(const v4f*)(xp);
    v4f xB = *(const v4f*)(xp + 4);
    v4f xC = *(const v4f*)(xp + 16);
    v4f xD = *(const v4f*)(xp + 20);

    // --- zero initial h (LDS) and c (registers, 2 cells per thread) ---
    hls[tid]       = (_Float16)0.0f;
    hls[tid + 256] = (_Float16)0.0f;
    float c0 = 0.0f, c1 = 0.0f;

    const int er = tid >> 5;                 // elementwise row 0..7 (and +8)
    const int ek = tid & 31;                 // elementwise hidden col
    float* outp0 = out + ((size_t)(b0 + er) * (size_t)T) * kH + ek;
    float* outp1 = out + ((size_t)(b0 + er + 8) * (size_t)T) * kH + ek;

    __syncthreads();

    for (int t = 0; t < T; ++t) {
        // h as A-matrix: two aligned 16B LDS reads per lane
        v16h Ah;
        {
            union { v16h v; v8h h[2]; } u;
            u.h[0] = *(const v8h*)(hls + col * kH + kb);
            u.h[1] = *(const v8h*)(hls + col * kH + kb + 16);
            Ah = u.v;
        }
        // x_t (prefetched last iteration) -> f16 A-matrix
        v16h Ax = cvt16(xA, xB, xC, xD);
        // prefetch x_{t+1} (independent of the recurrence: off critical path)
        if (t + 1 < T) {
            xp += kH;
            xA = *(const v4f*)(xp);
            xB = *(const v4f*)(xp + 4);
            xC = *(const v4f*)(xp + 16);
            xD = *(const v4f*)(xp + 20);
        }

        // gates = bias + x_t @ Wih^T + h @ Whh^T  (K=32 in ONE wmma each;
        // bias pre-loaded into the accumulator)
        v8f acc = __builtin_amdgcn_wmma_f32_16x16x32_f16(
                      false, Ax, false, Bih, (short)0, cbias, false, false);
        acc = __builtin_amdgcn_wmma_f32_16x16x32_f16(
                      false, Ah, false, Bhh, (short)0, acc, false, false);

        // scatter raw gate tile to LDS in C layout
#pragma unroll
        for (int v = 0; v < 8; ++v)
            Gls[(grow + v) * kG + n0 + col] = acc[v];
        __syncthreads();

        // elementwise LSTM cell: each thread owns 2 of the 16x32 cells
        {
            float iv = sigmoid_f(Gls[er * kG +       ek]);
            float fv = sigmoid_f(Gls[er * kG +  32 + ek]);
            float gv = tanh_f   (Gls[er * kG +  64 + ek]);
            float ov = sigmoid_f(Gls[er * kG +  96 + ek]);
            c0 = fv * c0 + iv * gv;
            float hv = ov * tanh_f(c0);
            outp0[(size_t)t * kH] = fminf(1.0f, fmaxf(-1.0f, hv));
            hls[er * kH + ek] = (_Float16)hv;
        }
        {
            const int r = er + 8;
            float iv = sigmoid_f(Gls[r * kG +       ek]);
            float fv = sigmoid_f(Gls[r * kG +  32 + ek]);
            float gv = tanh_f   (Gls[r * kG +  64 + ek]);
            float ov = sigmoid_f(Gls[r * kG +  96 + ek]);
            c1 = fv * c1 + iv * gv;
            float hv = ov * tanh_f(c1);
            outp1[(size_t)t * kH] = fminf(1.0f, fmaxf(-1.0f, hv));
            hls[r * kH + ek] = (_Float16)hv;
        }
        __syncthreads();
    }
}

extern "C" void kernel_launch(void* const* d_in, const int* in_sizes, int n_in,
                              void* d_out, int out_size, void* d_ws, size_t ws_size,
                              hipStream_t stream) {
    (void)n_in; (void)out_size; (void)d_ws; (void)ws_size;
    const float* x   = (const float*)d_in[0];
    const float* Wih = (const float*)d_in[1];
    const float* Whh = (const float*)d_in[2];
    const float* bih = (const float*)d_in[3];
    const float* bhh = (const float*)d_in[4];
    float* out = (float*)d_out;

    const int B = 256;
    const int T = in_sizes[0] / (B * kH);   // 2048 for the reference shapes

    dim3 grid(B / kMTile);   // 16 workgroups
    dim3 block(kThreads);    // 8 waves each
    hipLaunchKernelGGL(lstm_persistent_wmma, grid, block, 0, stream,
                       x, Wih, Whh, bih, bhh, out, T);
}
